// LSTMWithBias_27401891348776
// MI455X (gfx1250) — matrix operand order repaired and checked by reading, not compiled, not run on hardware
//
#include <hip/hip_runtime.h>

#define UNITS   256
#define IDIM    128
#define BATCHN  512
#define TSTEPS  256
#define GATES   1024          // 4*UNITS
#define KTOT    384           // IDIM + UNITS (combined GEMM K)
#define NTILES  64            // GATES / 16

typedef __attribute__((ext_vector_type(16))) __bf16 v16bf;
typedef __attribute__((ext_vector_type(8)))  float  v8f;
typedef __attribute__((ext_vector_type(4))) unsigned int u32x4;

// global (addrspace 1) pointer types so laundered loads stay global_load_*, not flat_load_*
typedef const unsigned short __attribute__((address_space(1)))* gus_p;
typedef const u32x4          __attribute__((address_space(1)))* gq4_p;

union AB32 {                  // one 16x32 bf16 WMMA operand fragment (8 VGPRs / lane)
    v16bf v;
    u32x4 q[2];
};

__device__ __forceinline__ unsigned short f2bf(float f) {
    unsigned u = __builtin_bit_cast(unsigned, f);
    unsigned r = u + 0x7FFFu + ((u >> 16) & 1u);   // round-to-nearest-even
    return (unsigned short)(r >> 16);
}

__device__ __forceinline__ float sigmoidf_fast(float x) {
    return 1.0f / (1.0f + __expf(-x));
}
__device__ __forceinline__ float tanhf_fast(float x) {
    return 1.0f - 2.0f / (__expf(2.0f * x) + 1.0f);
}

// ---------------- preprocessing kernels ----------------

// f32 -> packed bf16 pairs (x tensor, row-major [B,T,D])
__global__ void cvt_x_bf16(const float* __restrict__ x, unsigned int* __restrict__ xb, int n2) {
    int i = blockIdx.x * blockDim.x + threadIdx.x;
    if (i < n2) {
        float a = x[2 * i];
        float b = x[2 * i + 1];
        xb[i] = (unsigned)f2bf(a) | ((unsigned)f2bf(b) << 16);
    }
}

// Build weights in WMMA B-fragment order so the in-loop stream is fully coalesced:
//   wS[tile 0..63][kf 0..11][lane 0..31][j 0..15]
// holds W'(n = tile*16 + (lane&15), k = kf*32 + (lane>>4)*16 + j)
// where W'(n,k) = kernel[k][n] for k<128, rec_kernel[k-128][n] otherwise.
__global__ void build_wS(const float* __restrict__ k128, const float* __restrict__ r256,
                         unsigned short* __restrict__ wS) {
    int i = blockIdx.x * blockDim.x + threadIdx.x;   // output-linear
    if (i < GATES * KTOT) {
        int j    = i & 15;
        int f    = i >> 4;          // fragment-lane index
        int lane = f & 31;
        int fk   = f >> 5;
        int kf   = fk % 12;
        int tile = fk / 12;
        int n    = tile * 16 + (lane & 15);
        int k    = kf * 32 + (lane >> 4) * 16 + j;
        float v  = (k < IDIM) ? k128[(size_t)k * GATES + n]
                              : r256[(size_t)(k - IDIM) * GATES + n];
        wS[i] = f2bf(v);
    }
}

// ---------------- persistent LSTM kernel ----------------
// 32 workgroups x 512 threads (16 waves). WG b owns batch rows [16b, 16b+16).
// Step: g[16,1024] = [x_t | h][16,384] @ W' + bias  (bf16 WMMA, f32 acc), then gates.
__global__ __launch_bounds__(512, 1)
void lstm_persistent(const unsigned short* __restrict__ xb,   // [B,T,IDIM] bf16
                     const unsigned short* __restrict__ wS,   // fragment-ordered weights
                     const float* __restrict__ bias,          // [GATES] f32
                     float* __restrict__ hout)                // [B,UNITS] f32
{
    __shared__ unsigned short a_lds[16 * KTOT];   // 12 KB: [ x_t (0..127) | h (128..383) ]
    __shared__ float          g_lds[16 * GATES];  // 64 KB gate pre-activations
    __shared__ float          c_lds[16 * UNITS];  // 16 KB cell state

    const int tid  = threadIdx.x;
    const int wave = tid >> 5;
    const int lane = tid & 31;
    const int lo   = lane & 15;     // A: row | B/C: column
    const int hi   = lane >> 4;     // K sub-block / C row half
    const int r0   = blockIdx.x * 16;

    // zero cell state and h region of A
    for (int i = tid; i < 16 * UNITS; i += 512) {
        c_lds[i] = 0.0f;
        a_lds[(i >> 8) * KTOT + IDIM + (i & 255)] = 0;
    }
    // stage x_0 into A (1024 dwords, 2 per thread)
    const unsigned int* xw = (const unsigned int*)xb;
    unsigned int*       aw = (unsigned int*)a_lds;
#pragma unroll
    for (int j = 0; j < 2; ++j) {
        int idx = tid + j * 512;                 // 0..1023
        int row = idx >> 6, col = idx & 63;      // 64 dwords per row of x
        aw[row * (KTOT / 2) + col] = xw[((size_t)(r0 + row) * TSTEPS) * (IDIM / 2) + col];
    }
    __syncthreads();

    float biasv[4];
#pragma unroll
    for (int tI = 0; tI < 4; ++tI) biasv[tI] = bias[(wave * 4 + tI) * 16 + lo];

    for (int t = 0; t < TSTEPS; ++t) {
        // launder the (addrspace-1 typed) weight pointer: blocks LICM hoisting of the
        // per-step weight stream while keeping the loads as global_load_b128
        gus_p wSl = (gus_p)(unsigned long long)wS;
        asm volatile("" : "+s"(wSl));

        // ---- A fragments: [x_t | h] rows from LDS ----
        AB32 a[12];
        const unsigned short* ap = &a_lds[lo * KTOT + hi * 16];
#pragma unroll
        for (int kf = 0; kf < 12; ++kf) {
            a[kf].q[0] = *(const u32x4*)(ap + kf * 32);        // K = kf*32 + hi*16 .. +7
            a[kf].q[1] = *(const u32x4*)(ap + kf * 32 + 8);    // K = kf*32 + hi*16 + 8 .. +15
        }

        // ---- 4 N-tiles per wave (kept rolled to bound register pressure) ----
#pragma unroll 1
        for (int tI = 0; tI < 4; ++tI) {
            v8f acc;
#pragma unroll
            for (int r = 0; r < 8; ++r) acc[r] = biasv[tI];

            // fragment-ordered weights: fully coalesced, one base + immediate offsets
            // fragment kf lives at bp + kf*512 ushorts (1 KB)
            gus_p bp = wSl + (size_t)(wave * 4 + tI) * (12 * 32 * 16) + lane * 16;

            // 2-deep software-pipelined B stream: load frag kf+2 before WMMA kf
            AB32 b0, b1;
            b0.q[0] = *(gq4_p)(bp + 0);
            b0.q[1] = *(gq4_p)(bp + 8);
            b1.q[0] = *(gq4_p)(bp + 512);
            b1.q[1] = *(gq4_p)(bp + 520);
#pragma unroll
            for (int kf = 0; kf < 12; ++kf) {
                AB32 bn;
                if (kf + 2 < 12) {
                    bn.q[0] = *(gq4_p)(bp + (kf + 2) * 512);
                    bn.q[1] = *(gq4_p)(bp + (kf + 2) * 512 + 8);
                }
                acc = __builtin_amdgcn_wmma_f32_16x16x32_bf16(
                    false, a[kf].v, false, b0.v, (short)0, acc, false, false);
                b0 = b1;
                b1 = bn;
            }

            // C/D layout: reg r holds row (r + 8*hi), column = tile*16 + lo
            float* gp = &g_lds[(size_t)(8 * hi) * GATES + (wave * 4 + tI) * 16 + lo];
#pragma unroll
            for (int r = 0; r < 8; ++r) gp[(size_t)r * GATES] = acc[r];
        }
        __syncthreads();

        // ---- stage x_{t+1} into A's x region (writes ordered by the next barrier) ----
        if (t + 1 < TSTEPS) {
#pragma unroll
            for (int j = 0; j < 2; ++j) {
                int idx = tid + j * 512;
                int row = idx >> 6, col = idx & 63;
                aw[row * (KTOT / 2) + col] =
                    xw[((size_t)(r0 + row) * TSTEPS + (t + 1)) * (IDIM / 2) + col];
            }
        }

        // ---- gate nonlinearities + state update: 16*256 elems / 512 threads ----
#pragma unroll
        for (int e0 = 0; e0 < 8; ++e0) {
            int e  = e0 * 512 + tid;
            int mm = e >> 8;        // row 0..15
            int u  = e & 255;       // unit 0..255
            float gi = g_lds[mm * GATES + u];
            float gf = g_lds[mm * GATES + UNITS + u];
            float gc = g_lds[mm * GATES + 2 * UNITS + u];
            float go = g_lds[mm * GATES + 3 * UNITS + u];
            float i_ = sigmoidf_fast(gi);
            float f_ = sigmoidf_fast(gf);
            float cc = tanhf_fast(gc);
            float o_ = sigmoidf_fast(go);
            float cn = f_ * c_lds[e] + i_ * cc;
            float hn = o_ * tanhf_fast(cn);
            c_lds[e] = cn;
            a_lds[mm * KTOT + IDIM + u] = f2bf(hn);            // h for next step
            if (t == TSTEPS - 1) hout[(size_t)(r0 + mm) * UNITS + u] = hn;
        }
        __syncthreads();
    }
}

// ---------------- launcher ----------------
extern "C" void kernel_launch(void* const* d_in, const int* in_sizes, int n_in,
                              void* d_out, int out_size, void* d_ws, size_t ws_size,
                              hipStream_t stream) {
    const float* x      = (const float*)d_in[0];   // [512,256,128]
    const float* kernel = (const float*)d_in[1];   // [128,1024]
    const float* rec    = (const float*)d_in[2];   // [256,1024]
    const float* bias   = (const float*)d_in[3];   // [1024]
    float* hout = (float*)d_out;                   // [512,256]

    unsigned short* xb = (unsigned short*)d_ws;                   // 512*256*128 bf16 = 32 MB
    unsigned short* wS = xb + (size_t)BATCHN * TSTEPS * IDIM;     // fragment-ordered, 768 KB

    const int n2 = BATCHN * TSTEPS * IDIM / 2;
    cvt_x_bf16<<<(n2 + 255) / 256, 256, 0, stream>>>(x, (unsigned int*)xb, n2);
    build_wS<<<(GATES * KTOT + 255) / 256, 256, 0, stream>>>(kernel, rec, wS);

    lstm_persistent<<<BATCHN / 16, 512, 0, stream>>>(xb, wS, bias, hout);
}
